// MultiHeadAttention_24970939859328
// MI455X (gfx1250) — matrix operand order repaired and checked by reading, not compiled, not run on hardware
//
#include <hip/hip_runtime.h>

#define D_MODEL   1024
#define NUM_HEADS 16
#define D_K       64
#define BATCH     4
#define SEQ       2048
#define TOKENS    (BATCH * SEQ)   // 8192

typedef __attribute__((ext_vector_type(16))) __bf16 v16bf;
typedef __attribute__((ext_vector_type(8)))  __bf16 v8bf;
typedef __attribute__((ext_vector_type(8)))  float  v8f;
typedef __attribute__((ext_vector_type(4)))  int    v4i;

struct alignas(16) U4 { unsigned int x, y, z, w; };
struct alignas(16) F4 { float x, y, z, w; };

union FragU { v16bf v; U4 q[2]; };

// Native bf16 conversion for single values.
__device__ __forceinline__ unsigned short f2bf(float x) {
    return __builtin_bit_cast(unsigned short, (__bf16)x);
}

// 8 x fp32 -> 8 x bf16 as one vector convert: clean build_vector(fptrunc...)
// pattern for the backend to fuse into v_cvt_pk_bf16_f32 pairs; no integer
// repacking glue.
__device__ __forceinline__ U4 pack8(const F4& f0, const F4& f1) {
    v8f v;
    v[0] = f0.x; v[1] = f0.y; v[2] = f0.z; v[3] = f0.w;
    v[4] = f1.x; v[5] = f1.y; v[6] = f1.z; v[7] = f1.w;
    v8bf b = __builtin_convertvector(v, v8bf);
    return __builtin_bit_cast(U4, b);
}

// Load a 16-element bf16 fragment as two 16B vectors (ds_load_b128).
__device__ __forceinline__ v16bf load_frag(const unsigned short* p0,
                                           const unsigned short* p1) {
    FragU u;
    u.q[0] = *reinterpret_cast<const U4*>(p0);
    u.q[1] = *reinterpret_cast<const U4*>(p1);
    return u.v;
}

// ----- CDNA5 async global->LDS path (ASYNCcnt); proven live on this
// toolchain (round-2 diagnostic came from inside this guarded region). ------
#if __has_builtin(__builtin_amdgcn_global_load_async_to_lds_b128) && \
    __has_builtin(__builtin_amdgcn_s_wait_asynccnt)
#define USE_ASYNC_LDS 1
typedef __attribute__((address_space(1))) v4i ga_v4i;   // global b128 payload
typedef __attribute__((address_space(3))) v4i ls_v4i;   // LDS b128 payload
__device__ __forceinline__ void async_ld_b128(const void* g, const void* l) {
    __builtin_amdgcn_global_load_async_to_lds_b128(
        (ga_v4i*)(unsigned long long)g,
        (ls_v4i*)(unsigned int)(unsigned long long)l, 0, 0);
}
__device__ __forceinline__ void async_wait0() {
    __builtin_amdgcn_s_wait_asynccnt(0);
}
#else
#define USE_ASYNC_LDS 0
#endif

// ---------------------------------------------------------------------------
// Kernel 1: fused Q/K/V projection.  out = x @ W^T + b -> bf16 head-split
// [B][H][S][d_k].  Block tile 128x128, K tiles of 32, 8 waves of 32x64 each
// (8 WMMAs per K-step per wave).  Global->register prefetch pipelines the
// next K-tile under the WMMA block.
// ---------------------------------------------------------------------------
__global__ __launch_bounds__(256)
void qkv_proj_kernel(const float* __restrict__ q_in, const float* __restrict__ k_in,
                     const float* __restrict__ v_in,
                     const float* __restrict__ Wq, const float* __restrict__ bq,
                     const float* __restrict__ Wk, const float* __restrict__ bk,
                     const float* __restrict__ Wv, const float* __restrict__ bv,
                     unsigned short* __restrict__ qbuf,
                     unsigned short* __restrict__ kbuf,
                     unsigned short* __restrict__ vbuf)
{
    __shared__ __align__(16) unsigned short As[128 * 32];
    __shared__ __align__(16) unsigned short Bs[128 * 32];

    const int z = blockIdx.z;
    const float* x    = (z == 0) ? q_in : (z == 1) ? k_in : v_in;
    const float* W    = (z == 0) ? Wq   : (z == 1) ? Wk   : Wv;
    const float* bias = (z == 0) ? bq   : (z == 1) ? bk   : bv;
    unsigned short* out = (z == 0) ? qbuf : (z == 1) ? kbuf : vbuf;

    const int tid  = threadIdx.x;
    const int lane = tid & 31, wid = tid >> 5;
    const int waveM = wid >> 1, waveN = wid & 1;     // 4 x 2 waves
    const int tBase = blockIdx.x * 128;
    const int nBase = blockIdx.y * 128;

    v8f acc[2][4] = {};

    const int fr = tid >> 1;            // 0..127: row handled by this thread
    const int fc = (tid & 1) * 16;      // 0 or 16 within the 32-wide K tile

    F4 pa[4], pb[4];                    // prefetch registers (next K tile)
    auto load_regs = [&](int k0) {
        const float* ap = x + (size_t)(tBase + fr) * D_MODEL + k0 + fc;
        pa[0] = *reinterpret_cast<const F4*>(ap);
        pa[1] = *reinterpret_cast<const F4*>(ap + 4);
        pa[2] = *reinterpret_cast<const F4*>(ap + 8);
        pa[3] = *reinterpret_cast<const F4*>(ap + 12);
        const float* bp = W + (size_t)(nBase + fr) * D_MODEL + k0 + fc;
        pb[0] = *reinterpret_cast<const F4*>(bp);
        pb[1] = *reinterpret_cast<const F4*>(bp + 4);
        pb[2] = *reinterpret_cast<const F4*>(bp + 8);
        pb[3] = *reinterpret_cast<const F4*>(bp + 12);
    };
    load_regs(0);

    const int akb  = (lane >> 4) * 8;
    const int bcol = lane & 15;
    const int bkb  = (lane >> 4) * 8;

    for (int k0 = 0; k0 < D_MODEL; k0 += 32) {
        __syncthreads();
        *reinterpret_cast<U4*>(&As[fr * 32 + fc])     = pack8(pa[0], pa[1]);
        *reinterpret_cast<U4*>(&As[fr * 32 + fc + 8]) = pack8(pa[2], pa[3]);
        *reinterpret_cast<U4*>(&Bs[fr * 32 + fc])     = pack8(pb[0], pb[1]);
        *reinterpret_cast<U4*>(&Bs[fr * 32 + fc + 8]) = pack8(pb[2], pb[3]);
        __syncthreads();

        if (k0 + 32 < D_MODEL) load_regs(k0 + 32);   // overlaps the WMMAs below

        v16bf bfr[4];
#pragma unroll
        for (int j = 0; j < 4; ++j) {
            const unsigned short* bp2 = &Bs[(waveN * 64 + j * 16 + bcol) * 32 + bkb];
            bfr[j] = load_frag(bp2, bp2 + 16);
        }
#pragma unroll
        for (int mi = 0; mi < 2; ++mi) {
            const int arow = waveM * 32 + mi * 16 + (lane & 15);
            v16bf afrag = load_frag(&As[arow * 32 + akb], &As[arow * 32 + akb + 16]);
#pragma unroll
            for (int j = 0; j < 4; ++j)
                acc[mi][j] = __builtin_amdgcn_wmma_f32_16x16x32_bf16(
                    false, afrag, false, bfr[j], (short)0, acc[mi][j], false, false);
        }
    }

    // epilogue: bias, convert, scatter to [B][H][S][d_k]
    const int hi = lane >> 4;
#pragma unroll
    for (int mi = 0; mi < 2; ++mi)
#pragma unroll
        for (int j = 0; j < 4; ++j) {
            const int nG = nBase + waveN * 64 + j * 16 + (lane & 15);
            const float bb = bias[nG];
            const int h = nG >> 6, d = nG & 63;
#pragma unroll
            for (int r = 0; r < 8; ++r) {
                const int tok = tBase + waveM * 32 + mi * 16 + r + hi * 8;
                const int b = tok >> 11, s = tok & (SEQ - 1);
                out[((size_t)(b * NUM_HEADS + h) * SEQ + s) * D_K + d] =
                    f2bf(acc[mi][j][r] + bb);
            }
        }
}

// ---------------------------------------------------------------------------
// Kernel 2: flash-attention core.  One block = one (b,h); 8 waves x 16 query
// rows; streams 64-key blocks.  K tiles arrive via double-buffered
// GLOBAL_LOAD_ASYNC_TO_LDS (ASYNCcnt); V tiles are prefetched to registers
// and transposed into LDS for the B-fragment layout.
// 16 WMMAs per key block (8 QK^T + 8 PV).
// ---------------------------------------------------------------------------
__global__ __launch_bounds__(256)
void attn_kernel(const unsigned short* __restrict__ qbuf,
                 const unsigned short* __restrict__ kbuf,
                 const unsigned short* __restrict__ vbuf,
                 const unsigned char* __restrict__ mask,
                 unsigned short* __restrict__ ctx)
{
    __shared__ __align__(16) unsigned short Kt[2][64 * 64];   // [buf][key][d]
    __shared__ __align__(16) unsigned short Vt[64 * 64];      // [d][key]
    __shared__ __align__(16) unsigned short Ps[8 * 16 * 64];  // per-wave P

    const int tid  = threadIdx.x, lane = tid & 31, wid = tid >> 5;
    const int bh   = blockIdx.x;
    const int b    = bh >> 4, h = bh & 15;
    const int sBase = blockIdx.y * 128 + wid * 16;

    const unsigned short* qptr = qbuf + ((size_t)bh * SEQ + sBase) * D_K;
    const unsigned short* kptr = kbuf + (size_t)bh * SEQ * D_K;
    const unsigned short* vptr = vbuf + (size_t)bh * SEQ * D_K;

    // Q fragments: 16 rows x d_k=64 (two K-chunks), resident in registers
    const int arow = lane & 15;
    const int akb  = (lane >> 4) * 8;
    v16bf aq0 = load_frag(qptr + arow * 64 + akb,      qptr + arow * 64 + akb + 16);
    v16bf aq1 = load_frag(qptr + arow * 64 + 32 + akb, qptr + arow * 64 + 32 + akb + 16);

    v8f acc[4] = {};
    float m[8], l[8];
#pragma unroll
    for (int r = 0; r < 8; ++r) { m[r] = -3.0e38f; l[r] = 0.0f; }

    // tile-fill coordinates
    const int lkey = tid >> 2, ld0 = (tid & 3) * 16;   // K: 16 bf16 / thread
    const int vkey = tid & 63, vd0 = (tid >> 6) * 16;  // V: 16 bf16 / thread
    const int bcol = lane & 15;
    const int bkb  = (lane >> 4) * 8;
    unsigned short* ps = &Ps[wid * 16 * 64];

    U4 va0, va1;                       // V prefetch registers
#if !USE_ASYNC_LDS
    U4 ka0, ka1;                       // K prefetch registers (sync fallback)
#endif
    auto load_v_regs = [&](int kb) {
        const unsigned short* p = vptr + (size_t)(kb + vkey) * 64 + vd0;
        va0 = *reinterpret_cast<const U4*>(p);
        va1 = *reinterpret_cast<const U4*>(p + 8);
    };
#if !USE_ASYNC_LDS
    auto load_k_regs = [&](int kb) {
        const unsigned short* p = kptr + (size_t)(kb + lkey) * 64 + ld0;
        ka0 = *reinterpret_cast<const U4*>(p);
        ka1 = *reinterpret_cast<const U4*>(p + 8);
    };
    load_k_regs(0);
#else
    // issue DMA for tile 0
    async_ld_b128(kptr + (size_t)lkey * 64 + ld0,     &Kt[0][lkey * 64 + ld0]);
    async_ld_b128(kptr + (size_t)lkey * 64 + ld0 + 8, &Kt[0][lkey * 64 + ld0 + 8]);
#endif
    load_v_regs(0);

    for (int kb0 = 0; kb0 < SEQ; kb0 += 64) {
        const int buf = (kb0 >> 6) & 1;
        __syncthreads();   // A: previous iteration done reading Vt / writing Ps

        // V transpose into LDS from prefetch registers
        {
            const unsigned short* e0 = reinterpret_cast<const unsigned short*>(&va0);
            const unsigned short* e1 = reinterpret_cast<const unsigned short*>(&va1);
#pragma unroll
            for (int i = 0; i < 8; ++i) Vt[(vd0 + i) * 64 + vkey]     = e0[i];
#pragma unroll
            for (int i = 0; i < 8; ++i) Vt[(vd0 + 8 + i) * 64 + vkey] = e1[i];
        }
#if USE_ASYNC_LDS
        async_wait0();     // this tile's K DMA complete (s_wait_asynccnt 0)
#else
        *reinterpret_cast<U4*>(&Kt[buf][lkey * 64 + ld0])     = ka0;
        *reinterpret_cast<U4*>(&Kt[buf][lkey * 64 + ld0 + 8]) = ka1;
#endif
        __syncthreads();   // B: Kt[buf] + Vt visible to all waves

        if (kb0 + 64 < SEQ) {          // start next tile's movement now
#if USE_ASYNC_LDS
            const unsigned short* np = kptr + (size_t)(kb0 + 64 + lkey) * 64 + ld0;
            async_ld_b128(np,     &Kt[buf ^ 1][lkey * 64 + ld0]);
            async_ld_b128(np + 8, &Kt[buf ^ 1][lkey * 64 + ld0 + 8]);
#else
            load_k_regs(kb0 + 64);
#endif
            load_v_regs(kb0 + 64);
        }

        // ---- scores: Q(16x64) @ K^T(64x64) -> four 16x16 tiles --------------
        v8f sc[4] = {};
#pragma unroll
        for (int n = 0; n < 4; ++n) {
            const unsigned short* bp0 = &Kt[buf][(n * 16 + bcol) * 64 + bkb];
            v16bf b0 = load_frag(bp0, bp0 + 16);
            sc[n] = __builtin_amdgcn_wmma_f32_16x16x32_bf16(
                false, aq0, false, b0, (short)0, sc[n], false, false);
            const unsigned short* bp1 = &Kt[buf][(n * 16 + bcol) * 64 + 32 + bkb];
            v16bf b1 = load_frag(bp1, bp1 + 16);
            sc[n] = __builtin_amdgcn_wmma_f32_16x16x32_bf16(
                false, aq1, false, b1, (short)0, sc[n], false, false);
        }

        // scale 1/sqrt(d_k) and apply key mask
#pragma unroll
        for (int n = 0; n < 4; ++n) {
            const int keyG = kb0 + n * 16 + bcol;
            const float madd = mask[b * SEQ + keyG] ? 0.0f : -3.0e38f;
#pragma unroll
            for (int r = 0; r < 8; ++r) sc[n][r] = sc[n][r] * 0.125f + madd;
        }

        // ---- online softmax (row reductions across 16-lane halves) ----------
        float rm[8], rs[8], esc[8];
#pragma unroll
        for (int r = 0; r < 8; ++r)
            rm[r] = fmaxf(fmaxf(sc[0][r], sc[1][r]), fmaxf(sc[2][r], sc[3][r]));
#pragma unroll
        for (int off = 8; off >= 1; off >>= 1)
#pragma unroll
            for (int r = 0; r < 8; ++r) rm[r] = fmaxf(rm[r], __shfl_xor(rm[r], off, 32));
#pragma unroll
        for (int r = 0; r < 8; ++r) {
            const float mn = fmaxf(m[r], rm[r]);
            esc[r] = __expf(m[r] - mn);
            m[r] = mn;
            l[r] *= esc[r];
        }
#pragma unroll
        for (int n = 0; n < 4; ++n)
#pragma unroll
            for (int r = 0; r < 8; ++r) sc[n][r] = __expf(sc[n][r] - m[r]);
#pragma unroll
        for (int r = 0; r < 8; ++r)
            rs[r] = (sc[0][r] + sc[1][r]) + (sc[2][r] + sc[3][r]);
#pragma unroll
        for (int off = 8; off >= 1; off >>= 1)
#pragma unroll
            for (int r = 0; r < 8; ++r) rs[r] += __shfl_xor(rs[r], off, 32);
#pragma unroll
        for (int r = 0; r < 8; ++r) l[r] += rs[r];

#pragma unroll
        for (int j = 0; j < 4; ++j)
#pragma unroll
            for (int r = 0; r < 8; ++r) acc[j][r] *= esc[r];

        // spill P (C-layout) to per-wave LDS row-major 16x64 bf16
        {
            const int hi = lane >> 4;
#pragma unroll
            for (int n = 0; n < 4; ++n)
#pragma unroll
                for (int r = 0; r < 8; ++r)
                    ps[(r + hi * 8) * 64 + n * 16 + bcol] = f2bf(sc[n][r]);
        }
        __syncthreads();   // C: wave's P rows complete in LDS

        // ---- PV: acc(16x64) += P(16x64) @ V(64x64) --------------------------
#pragma unroll
        for (int c = 0; c < 2; ++c) {
            v16bf ap = load_frag(&ps[arow * 64 + c * 32 + akb],
                                 &ps[arow * 64 + c * 32 + akb + 16]);
#pragma unroll
            for (int j = 0; j < 4; ++j) {
                const unsigned short* bp = &Vt[(j * 16 + bcol) * 64 + c * 32 + bkb];
                v16bf bfrag = load_frag(bp, bp + 16);
                acc[j] = __builtin_amdgcn_wmma_f32_16x16x32_bf16(
                    false, ap, false, bfrag, (short)0, acc[j], false, false);
            }
        }
    }

    // finalize: normalize, write merged-head bf16 ctx [token][D_MODEL]
    const int hi = lane >> 4;
    float linv[8];
#pragma unroll
    for (int r = 0; r < 8; ++r) linv[r] = 1.0f / l[r];
#pragma unroll
    for (int j = 0; j < 4; ++j) {
        const int d = h * 64 + j * 16 + (lane & 15);
#pragma unroll
        for (int r = 0; r < 8; ++r) {
            const int s = sBase + r + hi * 8;
            ctx[((size_t)b * SEQ + s) * D_MODEL + d] = f2bf(acc[j][r] * linv[r]);
        }
    }
}

// ---------------------------------------------------------------------------
// Kernel 3: output projection.  out = ctx(bf16) @ Wo^T + bo, fp32 result.
// Same 128x128 tiling / pipelining as kernel 1 (A is already bf16).
// ---------------------------------------------------------------------------
__global__ __launch_bounds__(256)
void out_proj_kernel(const unsigned short* __restrict__ ctx,
                     const float* __restrict__ Wo, const float* __restrict__ bo,
                     float* __restrict__ out)
{
    __shared__ __align__(16) unsigned short As[128 * 32];
    __shared__ __align__(16) unsigned short Bs[128 * 32];

    const int tid  = threadIdx.x;
    const int lane = tid & 31, wid = tid >> 5;
    const int waveM = wid >> 1, waveN = wid & 1;
    const int tBase = blockIdx.x * 128;
    const int nBase = blockIdx.y * 128;

    v8f acc[2][4] = {};

    const int fr = tid >> 1;
    const int fc = (tid & 1) * 16;

    U4 ca0, ca1;            // A prefetch (bf16 passthrough)
    F4 pb[4];               // B prefetch (fp32 -> bf16)
    auto load_regs = [&](int k0) {
        const unsigned short* apn = ctx + (size_t)(tBase + fr) * D_MODEL + k0 + fc;
        ca0 = *reinterpret_cast<const U4*>(apn);
        ca1 = *reinterpret_cast<const U4*>(apn + 8);
        const float* bp = Wo + (size_t)(nBase + fr) * D_MODEL + k0 + fc;
        pb[0] = *reinterpret_cast<const F4*>(bp);
        pb[1] = *reinterpret_cast<const F4*>(bp + 4);
        pb[2] = *reinterpret_cast<const F4*>(bp + 8);
        pb[3] = *reinterpret_cast<const F4*>(bp + 12);
    };
    load_regs(0);

    const int akb  = (lane >> 4) * 8;
    const int bcol = lane & 15;
    const int bkb  = (lane >> 4) * 8;

    for (int k0 = 0; k0 < D_MODEL; k0 += 32) {
        __syncthreads();
        *reinterpret_cast<U4*>(&As[fr * 32 + fc])     = ca0;
        *reinterpret_cast<U4*>(&As[fr * 32 + fc + 8]) = ca1;
        *reinterpret_cast<U4*>(&Bs[fr * 32 + fc])     = pack8(pb[0], pb[1]);
        *reinterpret_cast<U4*>(&Bs[fr * 32 + fc + 8]) = pack8(pb[2], pb[3]);
        __syncthreads();

        if (k0 + 32 < D_MODEL) load_regs(k0 + 32);

        v16bf bfr[4];
#pragma unroll
        for (int j = 0; j < 4; ++j) {
            const unsigned short* bp2 = &Bs[(waveN * 64 + j * 16 + bcol) * 32 + bkb];
            bfr[j] = load_frag(bp2, bp2 + 16);
        }
#pragma unroll
        for (int mi = 0; mi < 2; ++mi) {
            const int arowg = waveM * 32 + mi * 16 + (lane & 15);
            v16bf afrag = load_frag(&As[arowg * 32 + akb], &As[arowg * 32 + akb + 16]);
#pragma unroll
            for (int j = 0; j < 4; ++j)
                acc[mi][j] = __builtin_amdgcn_wmma_f32_16x16x32_bf16(
                    false, afrag, false, bfr[j], (short)0, acc[mi][j], false, false);
        }
    }

    const int hi = lane >> 4;
#pragma unroll
    for (int mi = 0; mi < 2; ++mi)
#pragma unroll
        for (int j = 0; j < 4; ++j) {
            const int nG = nBase + waveN * 64 + j * 16 + (lane & 15);
            const float bb = bo[nG];
#pragma unroll
            for (int r = 0; r < 8; ++r) {
                const int tok = tBase + waveM * 32 + mi * 16 + r + hi * 8;
                out[(size_t)tok * D_MODEL + nG] = acc[mi][j][r] + bb;
            }
        }
}

// ---------------------------------------------------------------------------
extern "C" void kernel_launch(void* const* d_in, const int* in_sizes, int n_in,
                              void* d_out, int out_size, void* d_ws, size_t ws_size,
                              hipStream_t stream)
{
    (void)in_sizes; (void)n_in; (void)out_size; (void)ws_size;

    const float* query = (const float*)d_in[0];
    const float* key_  = (const float*)d_in[1];
    const float* value = (const float*)d_in[2];
    const unsigned char* mask = (const unsigned char*)d_in[3];
    const float* Wq = (const float*)d_in[4];
    const float* bq = (const float*)d_in[5];
    const float* Wk = (const float*)d_in[6];
    const float* bk = (const float*)d_in[7];
    const float* Wv = (const float*)d_in[8];
    const float* bv = (const float*)d_in[9];
    const float* Wo = (const float*)d_in[10];
    const float* bo = (const float*)d_in[11];

    const size_t elems = (size_t)TOKENS * D_MODEL;  // 8,388,608 per tensor
    unsigned short* qbuf = (unsigned short*)d_ws;   // bf16 workspace: 4x16MB
    unsigned short* kbuf = qbuf + elems;
    unsigned short* vbuf = kbuf + elems;
    unsigned short* ctxb = vbuf + elems;

    qkv_proj_kernel<<<dim3(TOKENS / 128, D_MODEL / 128, 3), 256, 0, stream>>>(
        query, key_, value, Wq, bq, Wk, bk, Wv, bv, qbuf, kbuf, vbuf);

    attn_kernel<<<dim3(BATCH * NUM_HEADS, SEQ / 128), 256, 0, stream>>>(
        qbuf, kbuf, vbuf, mask, ctxb);

    out_proj_kernel<<<dim3(TOKENS / 128, D_MODEL / 128), 256, 0, stream>>>(
        ctxb, Wo, bo, (float*)d_out);
}
